// Attention_10711648436709
// MI455X (gfx1250) — matrix operand order repaired and checked by reading, not compiled, not run on hardware
//
#include <hip/hip_runtime.h>

// Batch-axis-softmax attention, fused single pass, bf16 WMMA (gfx1250 / CDNA5).
// B=4,H=16,S=2048,D=64.  scale = sqrt(H) = 4 (folded into Q as 0.25).
// softmax is over the BATCH axis -> elementwise across the 4 per-batch score
// tiles -> no running max/rescale, no score materialization.

typedef __attribute__((ext_vector_type(16))) __bf16 v16bf;
typedef __attribute__((ext_vector_type(8)))  float  v8f;
typedef __attribute__((ext_vector_type(4)))  float  f32x4;

#define B_ 4
#define H_ 16
#define S_ 2048
#define D_ 64
#define LOG2E 1.44269504088896340736f

__device__ __forceinline__ v16bf cvt16(f32x4 a, f32x4 b, f32x4 c, f32x4 d, float s) {
  v16bf r;
  r[0]  = (__bf16)(a[0]*s); r[1]  = (__bf16)(a[1]*s);
  r[2]  = (__bf16)(a[2]*s); r[3]  = (__bf16)(a[3]*s);
  r[4]  = (__bf16)(b[0]*s); r[5]  = (__bf16)(b[1]*s);
  r[6]  = (__bf16)(b[2]*s); r[7]  = (__bf16)(b[3]*s);
  r[8]  = (__bf16)(c[0]*s); r[9]  = (__bf16)(c[1]*s);
  r[10] = (__bf16)(c[2]*s); r[11] = (__bf16)(c[3]*s);
  r[12] = (__bf16)(d[0]*s); r[13] = (__bf16)(d[1]*s);
  r[14] = (__bf16)(d[2]*s); r[15] = (__bf16)(d[3]*s);
  return r;
}

// Load two 8-float groups (p0[0..7], p1[0..7]) -> v16bf, scaled.
__device__ __forceinline__ v16bf load2x8_bf16(const float* p0, const float* p1, float s) {
  f32x4 a = *(const f32x4*)(p0);
  f32x4 b = *(const f32x4*)(p0 + 4);
  f32x4 c = *(const f32x4*)(p1);
  f32x4 d = *(const f32x4*)(p1 + 4);
  return cvt16(a, b, c, d, s);
}

__global__ __launch_bounds__(256) void
attn_batchsoftmax_wmma(const float* __restrict__ q,
                       const float* __restrict__ k,
                       const float* __restrict__ v,
                       float* __restrict__ out) {
  // Wave-private P bounce buffer (accumulator layout -> A-operand layout).
  // Stride 36 floats: 16B-aligned rows, bank-conflict-free reads/writes.
  __shared__ float plds[8][16][36];

  const int lane = threadIdx.x & 31;
  const int wave = threadIdx.x >> 5;
  const int l16  = lane & 15;
  const int g    = lane >> 4;          // lane group (0: lanes 0-15, 1: 16-31)
  const int h    = blockIdx.y;
  const int i0   = (blockIdx.x * 8 + wave) * 16;

  // ---- Q (A operand) held in registers for all 4 batches, pre-scaled by 1/4.
  // A operand (contiguous k order, same in B): group g holds contiguous K-block {16g..16g+15}.
  v16bf qa[B_][2];
#pragma unroll
  for (int b = 0; b < B_; ++b) {
    const float* qrow = q + ((size_t)(b * H_ + h) * S_ + (i0 + l16)) * D_;
#pragma unroll
    for (int c = 0; c < 2; ++c)
      qa[b][c] = load2x8_bf16(qrow + c * 32 + 16 * g,
                              qrow + c * 32 + 16 * g + 8, 0.25f);
  }

  // ---- Output accumulators: 4 batches x 4 d-subtiles of 16x16 f32.
  v8f o[B_][4];
#pragma unroll
  for (int b = 0; b < B_; ++b)
#pragma unroll
    for (int dt = 0; dt < 4; ++dt)
      o[b][dt] = (v8f)(0.0f);

  // ================= stream over keys/values, 32 at a time =================
#pragma unroll 1
  for (int j0 = 0; j0 < S_; j0 += 32) {
    // ---- scores S_b = (Q/4) K_b^T for this j-tile (2 subtiles of 16 cols).
    v8f s[B_][2];
#pragma unroll
    for (int b = 0; b < B_; ++b) {
      const float* kbase = k + (size_t)(b * H_ + h) * S_ * D_;
#pragma unroll
      for (int jt = 0; jt < 2; ++jt) {
        v8f acc = (v8f)(0.0f);
#pragma unroll
        for (int c = 0; c < 2; ++c) {
          // B operand (contiguous k order, as in Q): lane = column (key j), group g holds
          // contiguous K-block {16g..16g+15} -> one 16-float contiguous read.
          const float* krow = kbase + (size_t)(j0 + jt * 16 + l16) * D_
                                    + c * 32 + 16 * g;
          v16bf kb = load2x8_bf16(krow, krow + 8, 1.0f);
          acc = __builtin_amdgcn_wmma_f32_16x16x32_bf16(
              false, qa[b][c], false, kb, (short)0, acc, false, false);
        }
        s[b][jt] = acc;
      }
    }

    // ---- softmax over the BATCH axis: purely elementwise across the 4
    // identically-laid-out accumulators. Exact within this j-tile.
#pragma unroll
    for (int jt = 0; jt < 2; ++jt)
#pragma unroll
      for (int r = 0; r < 8; ++r) {
        float s0 = s[0][jt][r], s1 = s[1][jt][r];
        float s2 = s[2][jt][r], s3 = s[3][jt][r];
        float m  = fmaxf(fmaxf(s0, s1), fmaxf(s2, s3));
        float e0 = __builtin_amdgcn_exp2f((s0 - m) * LOG2E);
        float e1 = __builtin_amdgcn_exp2f((s1 - m) * LOG2E);
        float e2 = __builtin_amdgcn_exp2f((s2 - m) * LOG2E);
        float e3 = __builtin_amdgcn_exp2f((s3 - m) * LOG2E);
        float rz = __builtin_amdgcn_rcpf(e0 + e1 + e2 + e3);
        s[0][jt][r] = e0 * rz; s[1][jt][r] = e1 * rz;
        s[2][jt][r] = e2 * rz; s[3][jt][r] = e3 * rz;
      }

    // ---- O_b += P_b @ V_b. P goes through wave-private LDS to convert
    // accumulator layout (row = vgpr + 8g, col = l16) into A-operand layout.
    // LDS ops are in-order within a wave: no barrier needed.
#pragma unroll
    for (int b = 0; b < B_; ++b) {
#pragma unroll
      for (int jt = 0; jt < 2; ++jt)
#pragma unroll
        for (int r = 0; r < 8; ++r)
          plds[wave][r + 8 * g][jt * 16 + l16] = s[b][jt][r];

      v16bf pa = load2x8_bf16(&plds[wave][l16][16 * g],
                              &plds[wave][l16][16 * g + 8], 1.0f);

      const float* vb_base = v + ((size_t)(b * H_ + h) * S_ + j0 + 16 * g) * D_;
#pragma unroll
      for (int dt = 0; dt < 4; ++dt) {
        // B-operand: lane = column d, group g holds rows j = j0+16g+e.
        const float* vcol = vb_base + dt * 16 + l16;
        v16bf vb;
#pragma unroll
        for (int e = 0; e < 16; ++e)
          vb[e] = (__bf16)vcol[(size_t)e * D_];
        o[b][dt] = __builtin_amdgcn_wmma_f32_16x16x32_bf16(
            false, pa, false, vb, (short)0, o[b][dt], false, false);
      }
    }

    // ---- prefetch next j-tile of K and V into cache (global_prefetch_b8).
    if (j0 + 32 < S_) {
#pragma unroll
      for (int b = 0; b < B_; ++b) {
        const size_t nrow = (size_t)(b * H_ + h) * S_ + (j0 + 32 + l16);
        __builtin_prefetch(k + nrow * D_, 0, 3);
        __builtin_prefetch(v + nrow * D_, 0, 3);
      }
    }
  }

  // ---- write O (f32 accumulator layout -> [B,H,S,D]).
#pragma unroll
  for (int b = 0; b < B_; ++b)
#pragma unroll
    for (int dt = 0; dt < 4; ++dt)
#pragma unroll
      for (int r = 0; r < 8; ++r)
        out[((size_t)(b * H_ + h) * S_ + i0 + r + 8 * g) * D_ + dt * 16 + l16] =
            o[b][dt][r];
}

extern "C" void kernel_launch(void* const* d_in, const int* in_sizes, int n_in,
                              void* d_out, int out_size, void* d_ws, size_t ws_size,
                              hipStream_t stream) {
  const float* q = (const float*)d_in[0];
  const float* k = (const float*)d_in[1];
  const float* v = (const float*)d_in[2];
  // d_in[3] (mask) intentionally unused: reference discards masked_fill result.
  float* out = (float*)d_out;
  (void)in_sizes; (void)n_in; (void)out_size; (void)d_ws; (void)ws_size;

  dim3 grid(S_ / 128, H_);   // 16 x 16 workgroups, 8 waves each (wave32)
  attn_batchsoftmax_wmma<<<grid, 256, 0, stream>>>(q, k, v, out);
}
